// HetGNNLayer_37366215475384
// MI455X (gfx1250) — matrix operand order repaired and checked by reading, not compiled, not run on hardware
//
#include <hip/hip_runtime.h>
#include <hip/hip_bf16.h>
#include <math.h>

// ---------------- problem constants (match reference) ----------------
#define N_LNC 16384
#define N_MI  8192
#define IN_LNC 256
#define IN_MI  128
#define HH 8
#define CC 32
#define HC 256           // H*C
#define E_LL 524288
#define E_MM 262144
#define E_LM 262144
#define NEG_SLOPE 0.2f

typedef __attribute__((ext_vector_type(16))) __bf16 v16bf;
typedef __attribute__((ext_vector_type(8)))  __bf16 v8bf;
typedef __attribute__((ext_vector_type(8)))  float  v8f;

// ---------------- f32 -> bf16 conversion (plain) ----------------
__global__ void cvt_bf16_kernel(const float* __restrict__ in,
                                __bf16* __restrict__ out, int n) {
  int i = blockIdx.x * blockDim.x + threadIdx.x;
  if (i < n) out[i] = (__bf16)in[i];
}

// ---------------- f32 -> bf16 conversion + transpose for weights ---------
// W: [K, 256] row-major  ->  Wt: [256, K] row-major (i.e., column-major W)
template <int K>
__global__ void cvtT_bf16_kernel(const float* __restrict__ W,
                                 __bf16* __restrict__ Wt) {
  int t = blockIdx.x * blockDim.x + threadIdx.x;   // t = k*256 + n
  if (t < K * HC) {
    int k = t >> 8, n = t & 255;
    Wt[(size_t)n * K + k] = (__bf16)W[t];
  }
}

// ---------------- fill ----------------
__global__ void fill_kernel(float* __restrict__ p, float v, int n) {
  int i = blockIdx.x * blockDim.x + threadIdx.x;
  if (i < n) p[i] = v;
}

// ---------------- WMMA GEMM: Out[N,256] = X[N,K] @ W[K,256] ----------------
// One wave computes a 16x64 output strip (4 x 16x16 WMMA tiles sharing one A
// fragment per K-step).  X is bf16 [N,K]; W is pre-transposed bf16 [256,K] so
// each lane's B fragment (one column, two 8-K runs 16 apart) is two b128 loads.
template <int K>
__global__ __launch_bounds__(32) void gemm_wmma_bf16_kernel(
    const __bf16* __restrict__ Xb,    // [N, K]
    const __bf16* __restrict__ Wtb,   // [256, K]
    float* __restrict__ Out) {        // [N, 256]
  const int lane = threadIdx.x;              // 0..31
  const int m0 = blockIdx.y * 16;
  const int n0 = blockIdx.x * 64;
  const int r16 = lane & 15;
  const bool hi = lane >= 16;
  const int abase = hi ? 8 : 0;              // A: K offset per lane half
  const int bbase = hi ? 8 : 0;              // B: K offset per lane half
  const __bf16* xrow = Xb + (size_t)(m0 + r16) * K;
  const __bf16* wc0 = Wtb + (size_t)(n0 + r16) * K + bbase;  // nt stride: 16*K

  v8f c0 = {}, c1 = {}, c2 = {}, c3 = {};
#pragma unroll
  for (int k0 = 0; k0 < K; k0 += 32) {
    // A fragment: two contiguous 16B loads per lane
    v8bf alo = *(const v8bf*)(xrow + k0 + abase);
    v8bf ahi = *(const v8bf*)(xrow + k0 + abase + 16);
    v16bf a = __builtin_shufflevector(alo, ahi, 0, 1, 2, 3, 4, 5, 6, 7,
                                      8, 9, 10, 11, 12, 13, 14, 15);
#pragma unroll
    for (int nt = 0; nt < 4; ++nt) {
      // B fragment: one column, two 8-K runs (same K order as A) = two 16B loads
      const __bf16* wcol = wc0 + (size_t)nt * 16 * K + k0;
      v8bf blo = *(const v8bf*)(wcol);
      v8bf bhi = *(const v8bf*)(wcol + 16);
      v16bf b = __builtin_shufflevector(blo, bhi, 0, 1, 2, 3, 4, 5, 6, 7,
                                        8, 9, 10, 11, 12, 13, 14, 15);
      v8f& c = (nt == 0) ? c0 : (nt == 1) ? c1 : (nt == 2) ? c2 : c3;
      // (neg_a, A, neg_b, B, c_mod, C, reuse_a, reuse_b)
      c = __builtin_amdgcn_wmma_f32_16x16x32_bf16(false, a, false, b,
                                                  (short)0, c, false, false);
    }
  }
  const int mrow = m0 + (hi ? 8 : 0);
#pragma unroll
  for (int nt = 0; nt < 4; ++nt) {
    const v8f& c = (nt == 0) ? c0 : (nt == 1) ? c1 : (nt == 2) ? c2 : c3;
    const int col = n0 + nt * 16 + r16;
#pragma unroll
    for (int r = 0; r < 8; ++r)
      Out[(size_t)(mrow + r) * HC + col] = c[r];
  }
}

// ---------------- float atomic max via int/uint atomics ----------------
__device__ __forceinline__ void atomicMaxF(float* addr, float val) {
  if (val >= 0.0f) atomicMax((int*)addr, __float_as_int(val));
  else             atomicMin((unsigned int*)addr, __float_as_uint(val));
}

// ---------------- per-edge logits + segment max ----------------
__global__ __launch_bounds__(256) void edge_logits_kernel(
    const float* __restrict__ hl, const float* __restrict__ hr,
    const int* __restrict__ src, const int* __restrict__ dst,
    const float* __restrict__ att,      // [H*C]
    float* __restrict__ logits,         // [E*H]
    float* __restrict__ mbuf,           // [Nd*H], pre-filled -inf
    int E) {
  int e = (blockIdx.x * blockDim.x + threadIdx.x) >> 5;
  int lane = threadIdx.x & 31;
  if (e >= E) return;
  const int s = src[e], d = dst[e];
  const float* pl = hl + (size_t)s * HC;
  const float* pr = hr + (size_t)d * HC;
#pragma unroll
  for (int h = 0; h < HH; ++h) {
    int idx = h * CC + lane;
    float x = pl[idx] + pr[idx];
    x = (x > 0.0f) ? x : NEG_SLOPE * x;
    float v = x * att[idx];
#pragma unroll
    for (int off = 16; off; off >>= 1) v += __shfl_xor(v, off, 32);
    if (lane == 0) {
      logits[(size_t)e * HH + h] = v;
      atomicMaxF(&mbuf[(size_t)d * HH + h], v);
    }
  }
}

// ---------------- exp + segment sum (ex overwrites logits in place) -------
__global__ void edge_exp_kernel(float* __restrict__ logits,
                                const int* __restrict__ dst,
                                const float* __restrict__ mbuf,
                                float* __restrict__ denom, int E) {
  int t = blockIdx.x * blockDim.x + threadIdx.x;
  if (t >= E * HH) return;
  int e = t >> 3, h = t & 7;
  int d = dst[e];
  float v = __expf(logits[t] - mbuf[(size_t)d * HH + h]);
  logits[t] = v;                         // now holds ex
  atomicAdd(&denom[(size_t)d * HH + h], v);
}

// ---------------- alpha-weighted scatter aggregation ----------------
__global__ __launch_bounds__(256) void edge_aggr_kernel(
    const float* __restrict__ hl,
    const int* __restrict__ src, const int* __restrict__ dst,
    const float* __restrict__ ex, const float* __restrict__ denom,
    float* __restrict__ agg, int E) {
  int e = (blockIdx.x * blockDim.x + threadIdx.x) >> 5;
  int lane = threadIdx.x & 31;
  if (e >= E) return;
  const int s = src[e], d = dst[e];
  const float* pl = hl + (size_t)s * HC;
  float* pd = agg + (size_t)d * HC;
#pragma unroll
  for (int h = 0; h < HH; ++h) {
    float alpha = ex[(size_t)e * HH + h] /
                  (denom[(size_t)d * HH + h] + 1e-16f);
    atomicAdd(&pd[h * CC + lane], pl[h * CC + lane] * alpha);
  }
}

// ---------------- finalize ----------------
__global__ void add_bias_kernel(float* __restrict__ out,
                                const float* __restrict__ b, int total) {
  int t = blockIdx.x * blockDim.x + threadIdx.x;
  if (t < total) out[t] += b[t & (HC - 1)];
}

__global__ void mean_bias_kernel(float* __restrict__ out,
                                 const float* __restrict__ a0,
                                 const float* __restrict__ a1,
                                 const float* __restrict__ b0,
                                 const float* __restrict__ b1, int total) {
  int t = blockIdx.x * blockDim.x + threadIdx.x;
  if (t < total) {
    int j = t & (HC - 1);
    out[t] = 0.5f * ((a0[t] + b0[j]) + (a1[t] + b1[j]));
  }
}

// =====================================================================
static inline size_t alignup(size_t x) { return (x + 255) & ~(size_t)255; }

extern "C" void kernel_launch(void* const* d_in, const int* in_sizes, int n_in,
                              void* d_out, int out_size, void* d_ws, size_t ws_size,
                              hipStream_t stream) {
  (void)in_sizes; (void)n_in; (void)out_size; (void)ws_size;
  const float* x_lnc = (const float*)d_in[0];
  const float* x_mi  = (const float*)d_in[1];
  const int* ei_ll = (const int*)d_in[2];
  const int* ei_mm = (const int*)d_in[3];
  const int* ei_lm = (const int*)d_in[4];
  const float* Wl_ll = (const float*)d_in[5];
  const float* Wr_ll = (const float*)d_in[6];
  const float* att_ll = (const float*)d_in[7];
  const float* b_ll   = (const float*)d_in[8];
  const float* Wl_mm = (const float*)d_in[9];
  const float* Wr_mm = (const float*)d_in[10];
  const float* att_mm = (const float*)d_in[11];
  const float* b_mm   = (const float*)d_in[12];
  const float* Wl_lm = (const float*)d_in[13];
  const float* Wr_lm = (const float*)d_in[14];
  const float* att_lm = (const float*)d_in[15];
  const float* b_lm   = (const float*)d_in[16];

  float* out_ll = (float*)d_out;                       // [N_LNC*HC]
  float* out_mi = (float*)d_out + (size_t)N_LNC * HC;  // [N_MI*HC]

  // ---- workspace layout ----
  char* ws = (char*)d_ws; size_t off = 0;
  auto carve = [&](size_t bytes) { char* p = ws + off; off += alignup(bytes); return p; };
  __bf16* xl_b    = (__bf16*)carve((size_t)N_LNC * IN_LNC * 2);
  __bf16* xm_b    = (__bf16*)carve((size_t)N_MI * IN_MI * 2);
  __bf16* wl_ll_t = (__bf16*)carve((size_t)IN_LNC * HC * 2);   // transposed
  __bf16* wr_ll_t = (__bf16*)carve((size_t)IN_LNC * HC * 2);
  __bf16* wl_mm_t = (__bf16*)carve((size_t)IN_MI * HC * 2);
  __bf16* wr_mm_t = (__bf16*)carve((size_t)IN_MI * HC * 2);
  __bf16* wl_lm_t = (__bf16*)carve((size_t)IN_LNC * HC * 2);
  __bf16* wr_lm_t = (__bf16*)carve((size_t)IN_MI * HC * 2);
  float* hl_ll = (float*)carve((size_t)N_LNC * HC * 4);
  float* hr_ll = (float*)carve((size_t)N_LNC * HC * 4);
  float* hl_mm = (float*)carve((size_t)N_MI * HC * 4);
  float* hr_mm = (float*)carve((size_t)N_MI * HC * 4);
  float* hl_lm = (float*)carve((size_t)N_LNC * HC * 4);
  float* hr_lm = (float*)carve((size_t)N_MI * HC * 4);
  float* lg_ll = (float*)carve((size_t)E_LL * HH * 4);   // logits then ex
  float* lg_mm = (float*)carve((size_t)E_MM * HH * 4);
  float* lg_lm = (float*)carve((size_t)E_LM * HH * 4);
  float* m_ll = (float*)carve((size_t)N_LNC * HH * 4);
  float* d_ll = (float*)carve((size_t)N_LNC * HH * 4);
  float* m_mm = (float*)carve((size_t)N_MI * HH * 4);
  float* d_mm = (float*)carve((size_t)N_MI * HH * 4);
  float* m_lm = (float*)carve((size_t)N_MI * HH * 4);
  float* d_lm = (float*)carve((size_t)N_MI * HH * 4);
  float* agg_mm = (float*)carve((size_t)N_MI * HC * 4);
  float* agg_lm = (float*)carve((size_t)N_MI * HC * 4);

  // ---- 1) convert inputs to bf16; weights to transposed bf16 ----
  cvt_bf16_kernel<<<(N_LNC * IN_LNC) / 256, 256, 0, stream>>>(x_lnc, xl_b, N_LNC * IN_LNC);
  cvt_bf16_kernel<<<(N_MI * IN_MI) / 256, 256, 0, stream>>>(x_mi, xm_b, N_MI * IN_MI);
  cvtT_bf16_kernel<IN_LNC><<<(IN_LNC * HC) / 256, 256, 0, stream>>>(Wl_ll, wl_ll_t);
  cvtT_bf16_kernel<IN_LNC><<<(IN_LNC * HC) / 256, 256, 0, stream>>>(Wr_ll, wr_ll_t);
  cvtT_bf16_kernel<IN_MI ><<<(IN_MI * HC) / 256, 256, 0, stream>>>(Wl_mm, wl_mm_t);
  cvtT_bf16_kernel<IN_MI ><<<(IN_MI * HC) / 256, 256, 0, stream>>>(Wr_mm, wr_mm_t);
  cvtT_bf16_kernel<IN_LNC><<<(IN_LNC * HC) / 256, 256, 0, stream>>>(Wl_lm, wl_lm_t);
  cvtT_bf16_kernel<IN_MI ><<<(IN_MI * HC) / 256, 256, 0, stream>>>(Wr_lm, wr_lm_t);

  // ---- 2) WMMA GEMMs (grid: 64-wide N strips x 16-row M tiles) ----
  dim3 gL(HC / 64, N_LNC / 16), gM(HC / 64, N_MI / 16), blk(32);
  gemm_wmma_bf16_kernel<IN_LNC><<<gL, blk, 0, stream>>>(xl_b, wl_ll_t, hl_ll);
  gemm_wmma_bf16_kernel<IN_LNC><<<gL, blk, 0, stream>>>(xl_b, wr_ll_t, hr_ll);
  gemm_wmma_bf16_kernel<IN_MI ><<<gM, blk, 0, stream>>>(xm_b, wl_mm_t, hl_mm);
  gemm_wmma_bf16_kernel<IN_MI ><<<gM, blk, 0, stream>>>(xm_b, wr_mm_t, hr_mm);
  gemm_wmma_bf16_kernel<IN_LNC><<<gL, blk, 0, stream>>>(xl_b, wl_lm_t, hl_lm);
  gemm_wmma_bf16_kernel<IN_MI ><<<gM, blk, 0, stream>>>(xm_b, wr_lm_t, hr_lm);

  // ---- 3) init accumulators ----
  auto fill = [&](float* p, float v, int n) {
    fill_kernel<<<(n + 255) / 256, 256, 0, stream>>>(p, v, n);
  };
  fill(out_ll, 0.0f, N_LNC * HC);
  fill(agg_mm, 0.0f, N_MI * HC);
  fill(agg_lm, 0.0f, N_MI * HC);
  fill(m_ll, -INFINITY, N_LNC * HH);  fill(d_ll, 0.0f, N_LNC * HH);
  fill(m_mm, -INFINITY, N_MI * HH);   fill(d_mm, 0.0f, N_MI * HH);
  fill(m_lm, -INFINITY, N_MI * HH);   fill(d_lm, 0.0f, N_MI * HH);

  const int* src_ll = ei_ll;           const int* dst_ll = ei_ll + E_LL;
  const int* src_mm = ei_mm;           const int* dst_mm = ei_mm + E_MM;
  const int* src_lm = ei_lm;           const int* dst_lm = ei_lm + E_LM;

  // ---- 4) logits + segment max (wave per edge) ----
  edge_logits_kernel<<<E_LL / 8, 256, 0, stream>>>(hl_ll, hr_ll, src_ll, dst_ll,
                                                   att_ll, lg_ll, m_ll, E_LL);
  edge_logits_kernel<<<E_MM / 8, 256, 0, stream>>>(hl_mm, hr_mm, src_mm, dst_mm,
                                                   att_mm, lg_mm, m_mm, E_MM);
  edge_logits_kernel<<<E_LM / 8, 256, 0, stream>>>(hl_lm, hr_lm, src_lm, dst_lm,
                                                   att_lm, lg_lm, m_lm, E_LM);

  // ---- 5) exp + segment sum ----
  edge_exp_kernel<<<E_LL * HH / 256, 256, 0, stream>>>(lg_ll, dst_ll, m_ll, d_ll, E_LL);
  edge_exp_kernel<<<E_MM * HH / 256, 256, 0, stream>>>(lg_mm, dst_mm, m_mm, d_mm, E_MM);
  edge_exp_kernel<<<E_LM * HH / 256, 256, 0, stream>>>(lg_lm, dst_lm, m_lm, d_lm, E_LM);

  // ---- 6) alpha-weighted aggregation ----
  edge_aggr_kernel<<<E_LL / 8, 256, 0, stream>>>(hl_ll, src_ll, dst_ll,
                                                 lg_ll, d_ll, out_ll, E_LL);
  edge_aggr_kernel<<<E_MM / 8, 256, 0, stream>>>(hl_mm, src_mm, dst_mm,
                                                 lg_mm, d_mm, agg_mm, E_MM);
  edge_aggr_kernel<<<E_LM / 8, 256, 0, stream>>>(hl_lm, src_lm, dst_lm,
                                                 lg_lm, d_lm, agg_lm, E_LM);

  // ---- 7) finalize ----
  add_bias_kernel<<<(N_LNC * HC) / 256, 256, 0, stream>>>(out_ll, b_ll, N_LNC * HC);
  mean_bias_kernel<<<(N_MI * HC) / 256, 256, 0, stream>>>(out_mi, agg_mm, agg_lm,
                                                          b_mm, b_lm, N_MI * HC);
}
